// NeuronCircuit_45140106281445
// MI455X (gfx1250) — matrix operand order repaired and checked by reading, not compiled, not run on hardware
//
#include <hip/hip_runtime.h>

// ---------------------------------------------------------------------------
// NeuronCircuit for MI455X (gfx1250): routed MoE projections + flash attention
// GEMMs on v_wmma_f32_16x16x32_f16 (wave32); V staging via TDM tensor_load.
// ---------------------------------------------------------------------------

typedef __attribute__((ext_vector_type(16))) _Float16 v16h;
typedef __attribute__((ext_vector_type(8)))  float    v8f;

#define D_MODEL   1024
#define RANK      512
#define N_HEADS   8
#define D_HEAD    64
#define N_EXPERTS 32
#define TOPK_C    8
#define TOPK_E    4
#define BATCH     2
#define SEQ       1024
#define NTOK      (BATCH*SEQ)      // 2048
#define CAP       NTOK             // worst-case tokens per expert bucket

#define MOE_ROWS  16
#define KSTEP     32
#define COLBLK    512

#if defined(__has_builtin)
#  if __has_builtin(__builtin_amdgcn_tensor_load_to_lds)
#    define HAVE_TDM 1
#  endif
#endif
#ifndef HAVE_TDM
#  define HAVE_TDM 0
#endif

#if HAVE_TDM
typedef __attribute__((ext_vector_type(4))) unsigned tdm_v4u;
typedef __attribute__((ext_vector_type(8))) int      tdm_v8i;
typedef __attribute__((ext_vector_type(4))) int      tdm_v4i;

__device__ __forceinline__ unsigned lds_byte_addr(const void* p) {
  // generic -> LDS(AS3) cast; AS3 offsets are what the D# lds_addr field wants
  return (unsigned)(unsigned long long)
         (__attribute__((address_space(3))) const void*)p;
}
#endif

// Fragment-order swizzle helpers (ISA 7.12.2, 16-bit A 16x32 / B 32x16).
// A element (m, k):  lane = m + 16*((k>>3)&1), ee = (k&7) | ((k>>1)&8)
// B element (k, n16): lane = n16 + 16*((k>>3)&1), ee = (k&7) | ((k>>1)&8)
// Fragment load for lane L is then Buf[L*16 .. L*16+15] (32B contiguous).

// ---------------------------------------------------------------------------
// zero-init kernel (atomic-accumulate targets + counters)
// ---------------------------------------------------------------------------
__global__ void zero_f32(float* __restrict__ p, long n) {
  long i = (long)blockIdx.x * blockDim.x + threadIdx.x;
  long stride = (long)gridDim.x * blockDim.x;
  for (; i < n; i += stride) p[i] = 0.0f;
}

// ---------------------------------------------------------------------------
// Router: one wave per token; lane = expert. dot(x_t, W_e), top-k via
// shfl-xor argmax, softmax over the k scores, atomic append to expert bucket.
// ---------------------------------------------------------------------------
__global__ __launch_bounds__(32)
void router_topk(const float* __restrict__ xin, int din,
                 const float* __restrict__ Wr,          // [32, din]
                 int topk,
                 int*   __restrict__ counts,            // [32]
                 int*   __restrict__ btok,              // [32*CAP]
                 float* __restrict__ bw)                // [32*CAP]
{
  const int t    = blockIdx.x;
  const int lane = threadIdx.x;                          // expert id
  const float* xrow = xin + (size_t)t * din;
  const float* wrow = Wr  + (size_t)lane * din;

  float s = 0.0f;
  for (int d = 0; d < din; d += 4) {
    float4 xv = *(const float4*)(xrow + d);
    float4 wv = *(const float4*)(wrow + d);
    s += xv.x*wv.x + xv.y*wv.y + xv.z*wv.z + xv.w*wv.w;
  }

  float topv[TOPK_C];
  int   topi[TOPK_C];
  float cur = s;
  for (int k = 0; k < topk; k++) {
    float m  = cur;
    int   mi = lane;
    #pragma unroll
    for (int off = 16; off > 0; off >>= 1) {
      float om = __shfl_xor(m,  off, 32);
      int   oi = __shfl_xor(mi, off, 32);
      if (om > m || (om == m && oi < mi)) { m = om; mi = oi; }
    }
    topv[k] = m; topi[k] = mi;          // identical on all lanes
    if (lane == mi) cur = -3.4e38f;
  }

  float wsum = 0.0f, w[TOPK_C];
  for (int k = 0; k < topk; k++) { w[k] = __expf(topv[k] - topv[0]); wsum += w[k]; }
  for (int k = 0; k < topk; k++) w[k] /= wsum;

  if (lane < topk) {
    const int e   = topi[lane];
    const int pos = atomicAdd(&counts[e], 1);
    btok[e * CAP + pos] = t;
    bw  [e * CAP + pos] = w[lane];
  }
}

// ---------------------------------------------------------------------------
// MoE gather-GEMM: grid = (expert, 16-token row tile, 512-col block).
// Block = 256 threads (8 waves); each wave owns 64 output columns.
// A (16x32) and B (32x512) staged in LDS *pre-swizzled into WMMA fragment
// order*, so each lane's fragment is a contiguous 32B LDS read (ds_load_b128).
// ---------------------------------------------------------------------------
__global__ __launch_bounds__(256)
void moe_gemm(const float* __restrict__ xin, int din,
              const float* __restrict__ neurons, int dout,   // [32, din, dout]
              const int*   __restrict__ counts,
              const int*   __restrict__ btok,
              const float* __restrict__ bw,
              float* __restrict__ out)
{
  const int e  = blockIdx.x;
  const int rt = blockIdx.y;
  const int cb = blockIdx.z;
  const int cnt = counts[e];
  if (rt * MOE_ROWS >= cnt) return;

  __shared__ __align__(32) _Float16 Abuf[MOE_ROWS * KSTEP];   // frag-order 16x32
  __shared__ __align__(32) _Float16 Bbuf[KSTEP * COLBLK];     // frag-order 32 tiles
  __shared__ int      rtok[MOE_ROWS];
  __shared__ float    rw[MOE_ROWS];

  const int tid  = threadIdx.x;
  const int wave = tid >> 5;
  const int lane = tid & 31;

  if (tid < MOE_ROWS) {
    const int idx = rt * MOE_ROWS + tid;
    if (idx < cnt) { rtok[tid] = btok[e * CAP + idx]; rw[tid] = bw[e * CAP + idx]; }
    else           { rtok[tid] = btok[e * CAP + cnt - 1]; rw[tid] = 0.0f; } // pad
  }
  __syncthreads();

  v8f acc[4];
  #pragma unroll
  for (int i = 0; i < 4; i++)
    #pragma unroll
    for (int j = 0; j < 8; j++) acc[i][j] = 0.0f;

  const float* nbase = neurons + (size_t)e * din * dout + (size_t)cb * COLBLK;

  for (int k0 = 0; k0 < din; k0 += KSTEP) {
    // stage A swizzled: pairs (k2, k2+1) are LDS-adjacent (ee, ee+1)
    for (int i = tid; i < MOE_ROWS * KSTEP / 2; i += 256) {
      const int m  = i >> 4;
      const int k2 = (i & 15) << 1;
      const float* xr = xin + (size_t)rtok[m] * din + k0;
      const float f0 = xr[k2], f1 = xr[k2 + 1];
      const int ln = m + (((k2 >> 3) & 1) << 4);
      const int ee = (k2 & 7) | ((k2 >> 1) & 8);
      _Float16* dst = &Abuf[ln * 16 + ee];
      dst[0] = (_Float16)f0; dst[1] = (_Float16)f1;
    }
    // stage B swizzled: 32 col-tiles of 16; pairs (k2, k2+1) adjacent in LDS
    for (int i = tid; i < KSTEP * COLBLK / 2; i += 256) {
      const int c  = i & (COLBLK - 1);
      const int k2 = (i >> 9) << 1;
      const float f0 = nbase[(size_t)(k0 + k2)     * dout + c];
      const float f1 = nbase[(size_t)(k0 + k2 + 1) * dout + c];
      const int t  = c >> 4;
      const int ln = (c & 15) + (((k2 >> 3) & 1) << 4);
      const int ee = (k2 & 7) | ((k2 >> 1) & 8);
      _Float16* dst = &Bbuf[t * 512 + ln * 16 + ee];
      dst[0] = (_Float16)f0; dst[1] = (_Float16)f1;
    }
    if (k0 + KSTEP < din)   // warm L2 for the next B rows
      __builtin_prefetch(nbase + (size_t)(k0 + KSTEP) * dout + tid, 0, 0);
    __syncthreads();

    const v16h a = *(const v16h*)&Abuf[lane * 16];
    #pragma unroll
    for (int ct = 0; ct < 4; ct++) {
      const v16h b = *(const v16h*)&Bbuf[(wave * 4 + ct) * 512 + lane * 16];
      acc[ct] = __builtin_amdgcn_wmma_f32_16x16x32_f16(
          false, a, false, b, (short)0, acc[ct], false, false);
    }
    __syncthreads();
  }

  // weighted scatter-accumulate (C layout: VGPR r -> row r / r+8)
  #pragma unroll
  for (int ct = 0; ct < 4; ct++) {
    const int n = cb * COLBLK + wave * 64 + ct * 16 + (lane & 15);
    #pragma unroll
    for (int r = 0; r < 8; r++) {
      const int   row = r + ((lane >> 4) << 3);
      const float val = acc[ct][r] * rw[row];
      __hip_atomic_fetch_add(&out[(size_t)rtok[row] * dout + n], val,
                             __ATOMIC_RELAXED, __HIP_MEMORY_SCOPE_AGENT);
    }
  }
}

// ---------------------------------------------------------------------------
// Flash attention: one wave per (b, h, 16-query tile). Causal mask analytic.
// V block (32 keys x 64) DMA'd into LDS with the Tensor Data Mover, issued
// before the QK^T WMMAs and waited with s_wait_tensorcnt just before P.V.
// ---------------------------------------------------------------------------
__global__ __launch_bounds__(32)
void flash_attn(const float* __restrict__ Q,
                const float* __restrict__ K,
                const float* __restrict__ V,
                float* __restrict__ O)
{
  const int qt   = blockIdx.x;
  const int h    = blockIdx.y;
  const int b    = blockIdx.z;
  const int lane = threadIdx.x;
  const int q0   = qt * 16;
  const size_t base = (size_t)b * SEQ * RANK + (size_t)h * D_HEAD;

  __shared__ __align__(32) _Float16 Pbuf[16 * 32];   // frag-order P relay
  __shared__ __align__(16) float    Vbuf[32 * D_HEAD]; // row-major fp32 (TDM)

  // Q A-fragments, 1/sqrt(d_head) folded in
  v16h qa[2];
  {
    const int m  = lane & 15;
    const int kb = (lane >> 4) << 3;
    const float* qrow = Q + base + (size_t)(q0 + m) * RANK;
    #pragma unroll
    for (int f = 0; f < 2; f++)
      #pragma unroll
      for (int ee = 0; ee < 16; ee++) {
        const int kk = f * 32 + kb + ((ee >> 3) << 4) + (((ee >> 1) & 3) << 1) + (ee & 1);
        qa[f][ee] = (_Float16)(qrow[kk] * 0.125f);
      }
  }

#if HAVE_TDM
  // D# group1 (constant across iterations): data_size=4B, tile 64x32,
  // tensor dims 1<<20 (no clipping), dim0 stride = RANK.
  tdm_v8i g1;
  g1[0] = 0x20000;            // [17:16] data_size = 2 (4 bytes)
  g1[1] = 0;                  // tensor_dim0[15:0] of (1<<20)
  g1[2] = 0x10;               // tensor_dim0[31:16]; tensor_dim1[15:0]=0
  g1[3] = 0x10 | (D_HEAD << 16);  // tensor_dim1[31:16]; tile_dim0 = 64
  g1[4] = 32;                 // tile_dim1 = 32 rows; tile_dim2 = 0
  g1[5] = RANK;               // tensor_dim0_stride[31:0]
  g1[6] = 0;
  g1[7] = 0;
  tdm_v4i gz; gz[0] = 0; gz[1] = 0; gz[2] = 0; gz[3] = 0;
  const unsigned vlds = lds_byte_addr(&Vbuf[0]);
#endif

  v8f o[4];
  float mst[8], lst[8];
  #pragma unroll
  for (int i = 0; i < 4; i++)
    #pragma unroll
    for (int j = 0; j < 8; j++) o[i][j] = 0.0f;
  #pragma unroll
  for (int r = 0; r < 8; r++) { mst[r] = -1e30f; lst[r] = 0.0f; }

  const int kend = q0 + 16;                  // last valid key + 1
  for (int kb0 = 0; kb0 < kend; kb0 += 32) {
#if HAVE_TDM
    {   // issue the V-tile DMA now; it overlaps the QK^T WMMAs below
      const unsigned long long ga =
          (unsigned long long)(const void*)(V + base + (size_t)kb0 * RANK);
      tdm_v4u g0;
      g0[0] = 1u;                                   // count = 1 descriptor
      g0[1] = vlds;                                 // lds_addr
      g0[2] = (unsigned)ga;                         // global_addr[31:0]
      g0[3] = (unsigned)(ga >> 32) | 0x80000000u;   // addr[56:32] | type=2
#if __clang_major__ >= 23
      tdm_v8i g4; g4[0]=0; g4[1]=0; g4[2]=0; g4[3]=0; g4[4]=0; g4[5]=0; g4[6]=0; g4[7]=0;
      __builtin_amdgcn_tensor_load_to_lds(g0, g1, gz, gz, g4, 0);
#else
      __builtin_amdgcn_tensor_load_to_lds(g0, g1, gz, gz, 0);
#endif
    }
#else
    for (int i = lane; i < 32 * D_HEAD; i += 32)
      Vbuf[i] = V[base + (size_t)(kb0 + (i >> 6)) * RANK + (i & 63)];
#endif

    // ---- S = Q K^T for 32 keys (two 16x16 C tiles) ----
    v8f s[2];
    #pragma unroll
    for (int j = 0; j < 2; j++) {
      v8f c;
      #pragma unroll
      for (int jj = 0; jj < 8; jj++) c[jj] = 0.0f;
      const int key = kb0 + j * 16 + (lane & 15);
      const float* krow = K + base + (size_t)key * RANK;
      #pragma unroll
      for (int f = 0; f < 2; f++) {
        v16h kbf;
        const float* kp = krow + f * 32 + ((lane >> 4) << 3);
        #pragma unroll
        for (int q4 = 0; q4 < 4; q4++) {
          const float4 kv = *(const float4*)(kp + (q4 >> 1) * 16 + (q4 & 1) * 4);
          kbf[q4 * 4 + 0] = (_Float16)kv.x;
          kbf[q4 * 4 + 1] = (_Float16)kv.y;
          kbf[q4 * 4 + 2] = (_Float16)kv.z;
          kbf[q4 * 4 + 3] = (_Float16)kv.w;
        }
        c = __builtin_amdgcn_wmma_f32_16x16x32_f16(
            false, qa[f], false, kbf, (short)0, c, false, false);
      }
      #pragma unroll
      for (int r = 0; r < 8; r++) {
        const int row = q0 + r + ((lane >> 4) << 3);
        if (key > row) c[r] = -1e30f;       // causal mask
      }
      s[j] = c;
    }

    // ---- online softmax update (row stats live per 16-lane half) ----
    #pragma unroll
    for (int r = 0; r < 8; r++) {
      float v = fmaxf(s[0][r], s[1][r]);
      #pragma unroll
      for (int off = 8; off > 0; off >>= 1) v = fmaxf(v, __shfl_xor(v, off, 32));
      const float mnew  = fmaxf(mst[r], v);
      const float scale = __expf(mst[r] - mnew);
      mst[r] = mnew;
      s[0][r] = __expf(s[0][r] - mnew);
      s[1][r] = __expf(s[1][r] - mnew);
      float rs = s[0][r] + s[1][r];
      #pragma unroll
      for (int off = 8; off > 0; off >>= 1) rs += __shfl_xor(rs, off, 32);
      lst[r] = lst[r] * scale + rs;
      #pragma unroll
      for (int f = 0; f < 4; f++) o[f][r] *= scale;
    }

    // ---- write P to LDS already in A-fragment order ----
    #pragma unroll
    for (int j = 0; j < 2; j++)
      #pragma unroll
      for (int r = 0; r < 8; r++) {
        const int row = r + ((lane >> 4) << 3);
        const int col = j * 16 + (lane & 15);
        const int ln  = row + (((col >> 3) & 1) << 4);
        const int ee  = (col & 7) | ((col >> 1) & 8);
        Pbuf[ln * 16 + ee] = (_Float16)s[j][r];
      }
    __syncthreads();
#if HAVE_TDM
    __builtin_amdgcn_s_wait_tensorcnt(0);
#endif

    // ---- O += P V  (A = P 16x32, B = V 32x64 as 4 tiles) ----
    const v16h pa = *(const v16h*)&Pbuf[lane * 16];
    #pragma unroll
    for (int ct = 0; ct < 4; ct++) {
      v16h vb;
      const int c  = ct * 16 + (lane & 15);
      const int kb = (lane >> 4) << 3;
      #pragma unroll
      for (int ee = 0; ee < 16; ee++)
        vb[ee] = (_Float16)Vbuf[(kb + ee + ((ee >> 3) << 3)) * D_HEAD + c];
      o[ct] = __builtin_amdgcn_wmma_f32_16x16x32_f16(
          false, pa, false, vb, (short)0, o[ct], false, false);
    }
    __syncthreads();
  }

  // ---- epilogue: normalize by l, store ----
  #pragma unroll
  for (int ct = 0; ct < 4; ct++) {
    const int c = ct * 16 + (lane & 15);
    #pragma unroll
    for (int r = 0; r < 8; r++) {
      const int row = q0 + r + ((lane >> 4) << 3);
      O[base + (size_t)row * RANK + c] = o[ct][r] / lst[r];
    }
  }
}

// ---------------------------------------------------------------------------
// host launcher
// ---------------------------------------------------------------------------
extern "C" void kernel_launch(void* const* d_in, const int* in_sizes, int n_in,
                              void* d_out, int out_size, void* d_ws, size_t ws_size,
                              hipStream_t stream) {
  (void)in_sizes; (void)n_in; (void)out_size; (void)ws_size;
  const float* x  = (const float*)d_in[0];
  // d_in[1] is the causal mask; computed analytically in flash_attn.
  const float* cn = (const float*)d_in[2];   // [32, 1024, 512]
  const float* en = (const float*)d_in[3];   // [32, 512, 1024]
  const float* Wq = (const float*)d_in[4];
  const float* Wk = (const float*)d_in[5];
  const float* Wv = (const float*)d_in[6];
  const float* Wo = (const float*)d_in[7];
  float* out = (float*)d_out;

  char* ws = (char*)d_ws;
  size_t off = 0;
  float* Qb = (float*)(ws + off); off += (size_t)NTOK * RANK * sizeof(float);
  float* Kb = (float*)(ws + off); off += (size_t)NTOK * RANK * sizeof(float);
  float* Vb = (float*)(ws + off); off += (size_t)NTOK * RANK * sizeof(float);
  float* Ab = (float*)(ws + off); off += (size_t)NTOK * RANK * sizeof(float);
  int*   cnts = (int*)(ws + off);  off += (size_t)4 * N_EXPERTS * sizeof(int);
  int*   btok = (int*)(ws + off);  off += (size_t)4 * N_EXPERTS * CAP * sizeof(int);
  float* bwt  = (float*)(ws + off); off += (size_t)4 * N_EXPERTS * CAP * sizeof(float);

  // 0) zero atomic-accumulate targets + counters (every call: deterministic)
  zero_f32<<<2048, 256, 0, stream>>>(Qb, (long)3 * NTOK * RANK);   // Q,K,V contiguous
  zero_f32<<<2048, 256, 0, stream>>>(out, (long)NTOK * D_MODEL);
  zero_f32<<<1, 256, 0, stream>>>((float*)cnts, 4 * N_EXPERTS);

  // 1) compress routers (q=0, k=1, v=2)
  router_topk<<<NTOK, 32, 0, stream>>>(x, D_MODEL, Wq, TOPK_C,
      cnts + 0 * N_EXPERTS, btok + (size_t)0 * N_EXPERTS * CAP, bwt + (size_t)0 * N_EXPERTS * CAP);
  router_topk<<<NTOK, 32, 0, stream>>>(x, D_MODEL, Wk, TOPK_C,
      cnts + 1 * N_EXPERTS, btok + (size_t)1 * N_EXPERTS * CAP, bwt + (size_t)1 * N_EXPERTS * CAP);
  router_topk<<<NTOK, 32, 0, stream>>>(x, D_MODEL, Wv, TOPK_C,
      cnts + 2 * N_EXPERTS, btok + (size_t)2 * N_EXPERTS * CAP, bwt + (size_t)2 * N_EXPERTS * CAP);

  // 2) routed compress projections (d_model -> rank), WMMA gather-GEMM
  dim3 gC(N_EXPERTS, NTOK / MOE_ROWS, RANK / COLBLK);   // (32, 128, 1)
  moe_gemm<<<gC, 256, 0, stream>>>(x, D_MODEL, cn, RANK,
      cnts + 0 * N_EXPERTS, btok + (size_t)0 * N_EXPERTS * CAP, bwt + (size_t)0 * N_EXPERTS * CAP, Qb);
  moe_gemm<<<gC, 256, 0, stream>>>(x, D_MODEL, cn, RANK,
      cnts + 1 * N_EXPERTS, btok + (size_t)1 * N_EXPERTS * CAP, bwt + (size_t)1 * N_EXPERTS * CAP, Kb);
  moe_gemm<<<gC, 256, 0, stream>>>(x, D_MODEL, cn, RANK,
      cnts + 2 * N_EXPERTS, btok + (size_t)2 * N_EXPERTS * CAP, bwt + (size_t)2 * N_EXPERTS * CAP, Vb);

  // 3) causal flash attention, 8 heads x 64
  dim3 gA(SEQ / 16, N_HEADS, BATCH);                    // (64, 8, 2)
  flash_attn<<<gA, 32, 0, stream>>>(Qb, Kb, Vb, Ab);

  // 4) expand router (o = 3) over attention output
  router_topk<<<NTOK, 32, 0, stream>>>(Ab, RANK, Wo, TOPK_E,
      cnts + 3 * N_EXPERTS, btok + (size_t)3 * N_EXPERTS * CAP, bwt + (size_t)3 * N_EXPERTS * CAP);

  // 5) routed expand projection (rank -> d_model) into d_out
  dim3 gE(N_EXPERTS, NTOK / MOE_ROWS, D_MODEL / COLBLK); // (32, 128, 2)
  moe_gemm<<<gE, 256, 0, stream>>>(Ab, RANK, en, D_MODEL,
      cnts + 3 * N_EXPERTS, btok + (size_t)3 * N_EXPERTS * CAP, bwt + (size_t)3 * N_EXPERTS * CAP, out);
}
